// diff_pred_19301583028960
// MI455X (gfx1250) — compile-verified, matrix operand order checked by reading, not run on hardware
//
#include <hip/hip_runtime.h>
#include <hip/hip_bf16.h>

// LPC prediction: out[b,t] = -sum_{i<16} lpc[b, t/160, i] * x[b, t-i]  (x[t<0]=0)
// B=1024, T=2400, N=16, F=160.
//
// Memory-bound (~20.6 MB @ 23.3 TB/s -> ~0.9 us floor). One wave32 per frame:
//   - stage x window [t0-15, t0+160) + 16 coeffs in LDS
//   - 10 output tiles of 16 samples, each one v_wmma_f32_16x16x32_f16
//     (A = 16x16 Toeplitz of x, K zero-padded to 32; B = coeff column-broadcast)
//   - coalesced negated store of 160 outputs.
//
// K is permuted pairwise (e^1) identically in A and B so A's half-pairs ascend
// in LDS address: ds_load_2addr + v_cvt_pk_f16_f32 pack fragments directly.

typedef _Float16 half16 __attribute__((ext_vector_type(16)));
typedef float    float8 __attribute__((ext_vector_type(8)));

#define B_  1024
#define T_  2400
#define N_  16
#define F_  160
#define FPB (T_ / F_)            // 15 frames per batch
#define TOTAL_FRAMES (B_ * FPB)  // 15360
#define WAVES_PER_BLOCK 4
#define WIN 176                  // 15 pre-pad + 160 samples, rounded to 176

__global__ __launch_bounds__(WAVES_PER_BLOCK * 32)
void lpc_pred_wmma(const float* __restrict__ xt,
                   const float* __restrict__ lpc,
                   float* __restrict__ out)
{
    __shared__ float s_x[WAVES_PER_BLOCK][WIN];   // signal window per wave
    __shared__ float s_l[WAVES_PER_BLOCK][N_];    // coeffs per wave
    __shared__ float s_o[WAVES_PER_BLOCK][F_];    // staged outputs per wave

    const int lane = threadIdx.x & 31;
    const int w    = threadIdx.x >> 5;
    const int gf   = blockIdx.x * WAVES_PER_BLOCK + w;   // global frame id
    const int b    = gf / FPB;
    const int f    = gf % FPB;
    const int t0   = f * F_;

    // ---- Cooperative load: x[t0-15 .. t0+159] (zeros before t=0), coeffs ----
    const float* xb = xt + b * T_;
    #pragma unroll
    for (int j = lane; j < WIN; j += 32) {
        const int idx = t0 - (N_ - 1) + j;
        float v = 0.0f;
        if (idx >= 0 && idx < T_) v = xb[idx];
        s_x[w][j] = v;
    }
    if (lane < N_) s_l[w][lane] = lpc[(b * FPB + f) * N_ + lane];
    __syncthreads();

    const int m = lane & 15;   // row / column index within 16x16 tile
    const int h = lane >> 4;   // lane half selects K sub-range

    // ---- B fragment: 32x16, column-broadcast of L[k(e)]; K in [16,32) zero --
    // 16-bit B layout (mirrors A): lanes 0-15 carry K sub-range 0..7, lanes
    // 16-31 K 8..15 in elements 0..7; elements 8..15 map to K>=16 (padding).
    // K permutation k(e) = 8h + (e^1), applied identically to A and B.
    half16 bfrag;
    #pragma unroll
    for (int e = 0; e < 8; ++e) bfrag[e] = (_Float16)s_l[w][8 * h + (e ^ 1)];
    #pragma unroll
    for (int e = 8; e < 16; ++e) bfrag[e] = (_Float16)0.0f;

    // ---- 10 tiles of 16 outputs: D[m,n] = sum_k x[ts+m-k] * L[k] ----
    #pragma unroll
    for (int tt = 0; tt < 10; ++tt) {
        const int base = (N_ - 1) + 16 * tt;   // LDS index of tile start ts

        half16 a;
        #pragma unroll
        for (int e = 0; e < 8; ++e)            // A[m, k(e)] = x[ts+m-k(e)]
            a[e] = (_Float16)s_x[w][base + m - (8 * h + (e ^ 1))];
        #pragma unroll
        for (int e = 8; e < 16; ++e) a[e] = (_Float16)0.0f;    // K >= 16 pad

        float8 c = {};
        float8 d = __builtin_amdgcn_wmma_f32_16x16x32_f16(
            /*neg_a=*/false, a, /*neg_b=*/false, bfrag,
            /*c_mod=*/(short)0, c, /*reuse_a=*/false, /*reuse_b=*/false);

        // All D columns are identical (B is column-broadcast): lanes 0 and 16
        // hold tile rows 8h+0..8h+7 in d[0..7]. WMMA is convergent and executes
        // before this branch, so EXEC is all-1s at the matrix op.
        if (m == 0) {
            #pragma unroll
            for (int r = 0; r < 8; ++r)
                s_o[w][16 * tt + 8 * h + r] = d[r];   // 32B-aligned contiguous
        }
    }
    __syncthreads();

    // ---- Coalesced negated write-back of the 160 frame outputs ----
    float* ob = out + b * T_ + t0;
    #pragma unroll
    for (int j = lane; j < F_; j += 32) ob[j] = -s_o[w][j];
}

extern "C" void kernel_launch(void* const* d_in, const int* in_sizes, int n_in,
                              void* d_out, int out_size, void* d_ws, size_t ws_size,
                              hipStream_t stream) {
    (void)in_sizes; (void)n_in; (void)d_ws; (void)ws_size; (void)out_size;
    const float* xt  = (const float*)d_in[0];   // [1024, 2400, 1]
    const float* lpc = (const float*)d_in[1];   // [1024, 15, 16]
    // d_in[2]=lpcoeffs_N(16), d_in[3]=frame_size(160): compile-time constants.
    float* out = (float*)d_out;                 // [1024, 2400, 1]

    dim3 grid(TOTAL_FRAMES / WAVES_PER_BLOCK);  // 3840 blocks
    dim3 block(WAVES_PER_BLOCK * 32);           // 4 waves of 32
    lpc_pred_wmma<<<grid, block, 0, stream>>>(xt, lpc, out);
}
